// CrossAttentionFusion_25142738551157
// MI455X (gfx1250) — matrix operand order repaired and checked by reading, not compiled, not run on hardware
//
#include <hip/hip_runtime.h>
#include <hip/hip_bf16.h>

// ---------------------------------------------------------------------------
// Cross-attention fusion for MI455X (gfx1250, wave32, WMMA + async-LDS).
// B=2, D=256, H=W=64 (P=4096 pixels), NH=4 heads, hd=64.
//
//   1) convert 1x1-conv weights f32->f16
//   2) per-pixel channel LayerNorm of x and guide -> f16 (b*P, 256)
//   3) q/k/v 1x1 convs as WMMA f16 GEMMs -> f32 NCHW scratch
//   4) depthwise 3x3 + bias (+ q scale 1/8) -> f16 head layouts
//        Q,K: (b*NH, P, 64)   V: transposed (b*NH, 64, P)
//   5) flash attention: 8 waves/block share one head; K/V j-chunks are
//      double-buffered into LDS with global_load_async_to_lds_b128
//      (ASYNCcnt / s_wait_asynccnt), S^T = K*Q^T trick for cheap softmax,
//      O^T = V^T * P^T; output f16 (b, P, 256)
//   6) output projection GEMM + bias + residual -> d_out f32 NCHW
// ---------------------------------------------------------------------------

typedef __attribute__((ext_vector_type(16))) _Float16 v16h;
typedef __attribute__((ext_vector_type(8)))  float    v8f;

#define D_C 256
#define PIX 4096
#define N_H 4
#define H_D 64
#define IMG_W 64

union F16x16 { v16h v; uint4 u4[2]; };
union F16x8  { _Float16 h[8]; uint4 u; };

__device__ __forceinline__ v8f wmma_f16(const v16h& a, const v16h& b, const v8f& c) {
    return __builtin_amdgcn_wmma_f32_16x16x32_f16(false, a, false, b, (short)0, c, false, false);
}

// Async global->LDS copy, 16B per lane. LDS operand is the wave-relative byte
// offset (low 32 bits of the generic shared-memory address per the LDS
// aperture rule addr[31:0] -> LDS_ADDR). Tracked by ASYNCcnt.
__device__ __forceinline__ void async_ld_b128(void* lds_ptr, const void* gaddr) {
    uint32_t lds_off = (uint32_t)(uintptr_t)lds_ptr;
    asm volatile("global_load_async_to_lds_b128 %0, %1, off"
                 :: "v"(lds_off), "v"(gaddr) : "memory");
}
__device__ __forceinline__ void wait_async_le2() {
    asm volatile("s_wait_asynccnt 2" ::: "memory");
}
__device__ __forceinline__ void wait_async_le0() {
    asm volatile("s_wait_asynccnt 0" ::: "memory");
}

// ---------------------------------------------------------------- weights cvt
__global__ __launch_bounds__(256) void cvt_f16_kernel(const float* __restrict__ in,
                                                      _Float16* __restrict__ out, int n) {
    int i = blockIdx.x * 256 + threadIdx.x;
    if (i < n) out[i] = (_Float16)in[i];
}

// ------------------------------------------------------------------ layernorm
__global__ __launch_bounds__(256) void lnorm_kernel(const float* __restrict__ X,   // (B,256,P)
                                                    const float* __restrict__ gam,
                                                    const float* __restrict__ bet,
                                                    _Float16* __restrict__ Out) {  // (B*P,256)
    const int lane = threadIdx.x & 31;
    const int wave = threadIdx.x >> 5;
    const int gp   = blockIdx.x * 8 + wave;          // 0..B*P-1
    const int b    = gp >> 12;
    const int p    = gp & (PIX - 1);
    const float* base = X + (size_t)b * D_C * PIX + p;

    float vals[8];
    float s = 0.f;
#pragma unroll
    for (int i = 0; i < 8; ++i) { vals[i] = base[(size_t)(lane + 32 * i) * PIX]; s += vals[i]; }
#pragma unroll
    for (int m = 16; m >= 1; m >>= 1) s += __shfl_xor(s, m, 32);
    const float mu = s * (1.0f / D_C);
    float v = 0.f;
#pragma unroll
    for (int i = 0; i < 8; ++i) { float d = vals[i] - mu; v += d * d; }
#pragma unroll
    for (int m = 16; m >= 1; m >>= 1) v += __shfl_xor(v, m, 32);
    const float rs = rsqrtf(v * (1.0f / D_C) + 1e-5f);

    _Float16* out = Out + (size_t)gp * D_C;
#pragma unroll
    for (int i = 0; i < 8; ++i) {
        int c = lane + 32 * i;
        out[c] = (_Float16)((vals[i] - mu) * rs * gam[c] + bet[c]);
    }
}

// --------------------------------------------------------------- 1x1 conv GEMM
__global__ __launch_bounds__(256) void gemm1x1_kernel(const _Float16* __restrict__ A,  // (B*P,256)
                                                      const _Float16* __restrict__ Wf, // (256,256)
                                                      const float* __restrict__ bias,
                                                      const float* __restrict__ resid, // NCHW or null
                                                      float* __restrict__ Y) {         // (B,256,P)
    const int lane = threadIdx.x & 31;
    const int wave = threadIdx.x >> 5;
    const int n = lane & 15, h = lane >> 4;
    const int gp0 = blockIdx.x * 16;                 // pixel-row tile base
    const int o0  = (blockIdx.y * 8 + wave) * 16;    // out-channel tile base

    v8f acc = {0, 0, 0, 0, 0, 0, 0, 0};
    const _Float16* arow = A  + (size_t)(gp0 + n) * D_C;   // A row m = lane&15
    const _Float16* brow = Wf + (size_t)(o0  + n) * D_C;   // B col n = out channel
#pragma unroll
    for (int kk = 0; kk < D_C; kk += 32) {
        F16x16 af, bf;
        af.u4[0] = *(const uint4*)(arow + kk + 8 * h);        // k = kk+8h..+7
        af.u4[1] = *(const uint4*)(arow + kk + 16 + 8 * h);   // k = kk+16+8h..+7
        bf.u4[0] = *(const uint4*)(brow + kk + 8 * h);        // k = kk+8h..+7
        bf.u4[1] = *(const uint4*)(brow + kk + 16 + 8 * h);
        acc = wmma_f16(af.v, bf.v, acc);
    }
    const int o = o0 + n;
    const float bv = bias[o];
#pragma unroll
    for (int r = 0; r < 8; ++r) {
        int gp = gp0 + 8 * h + r;                    // C row m = 8h+r
        int bb = gp >> 12, p = gp & (PIX - 1);
        size_t idx = ((size_t)bb * D_C + o) * PIX + p;
        float val = acc[r] + bv;
        if (resid) val += resid[idx];
        Y[idx] = val;
    }
}

// ------------------------------------------------------- depthwise 3x3 + pack
// layoutT==0 : Out[(b*NH+h), p, d]  (Q, K)   layoutT==1 : Out[(b*NH+h), d, p] (V^T)
__global__ __launch_bounds__(256) void dwconv_kernel(const float* __restrict__ Y,  // (B,256,P)
                                                     const float* __restrict__ w,  // (256,3,3)
                                                     const float* __restrict__ bias,
                                                     float scale, int layoutT,
                                                     _Float16* __restrict__ Out) {
    int idx = blockIdx.x * 256 + threadIdx.x;        // over B*256*P
    int p  = idx & (PIX - 1);
    int co = (idx >> 12) & (D_C - 1);
    int b  = idx >> 20;
    int y = p >> 6, x = p & (IMG_W - 1);
    const float* src = Y + ((size_t)b * D_C + co) * PIX;
    const float* wc  = w + co * 9;
    float acc = bias[co];
#pragma unroll
    for (int dy = -1; dy <= 1; ++dy)
#pragma unroll
        for (int dx = -1; dx <= 1; ++dx) {
            int yy = y + dy, xx = x + dx;
            if (yy >= 0 && yy < IMG_W && xx >= 0 && xx < IMG_W)
                acc += wc[(dy + 1) * 3 + (dx + 1)] * src[yy * IMG_W + xx];
        }
    acc *= scale;
    int hh = co >> 6, d = co & (H_D - 1);
    size_t o = layoutT ? (((size_t)(b * N_H + hh) * H_D + d) * PIX + p)
                       : (((size_t)(b * N_H + hh) * PIX + p) * H_D + d);
    Out[o] = (_Float16)acc;
}

// --------------------------------------------------------------- attention
// Block = 8 waves on 8 consecutive 16-query tiles of ONE (b,head). K/V key
// chunks (32 keys) are async-copied into double-buffered LDS and shared by
// all waves. Per wave: S^T = K*Q^T (query index lands in the lane dim of the
// C fragment -> softmax = own 8 values + one shfl_xor(16)); P^T B-fragment
// rebuilt from S^T C-fragments with half-wave exchanges; O^T += V^T * P^T.
__global__ __launch_bounds__(256) void attn_kernel(const _Float16* __restrict__ Qf, // (BH,P,64)
                                                   const _Float16* __restrict__ Kf, // (BH,P,64)
                                                   const _Float16* __restrict__ Vt, // (BH,64,P)
                                                   _Float16* __restrict__ Of) {     // (B,P,256)
    __shared__ _Float16 smK[2][32 * H_D];   // 32 keys x 64 d   (4 KB each)
    __shared__ _Float16 smV[2][H_D * 32];   // 64 d  x 32 keys  (4 KB each)

    const int tid  = threadIdx.x;
    const int lane = tid & 31;
    const int wave = tid >> 5;
    const int n = lane & 15, h = lane >> 4;

    const int bh = blockIdx.x >> 5;                   // 32 blocks per (b,head)
    const int i0 = (((blockIdx.x & 31) << 3) + wave) * 16;
    const int b  = bh >> 2, hh = bh & 3;

    const _Float16* Qp = Qf + (size_t)bh * PIX * H_D;
    const _Float16* Kp = Kf + (size_t)bh * PIX * H_D;
    const _Float16* Vp = Vt + (size_t)bh * H_D * PIX;

    // staging coordinates for this thread (16B segment per tensor per stage)
    const int krow = tid >> 3, kseg = (tid & 7) * 8;  // K: 32 rows x 128B
    const int vrow = tid >> 2, vseg = (tid & 3) * 8;  // V: 64 rows x 64B

    // Q as B-fragments (loaded once): b[e] = Q[i0+n][dbase + 8h + e], b[e+8] = Q[i0+n][dbase + 16 + 8h + e]
    F16x16 qb0, qb1;
    {
        const _Float16* q = Qp + (size_t)(i0 + n) * H_D + 8 * h;
        qb0.u4[0] = *(const uint4*)(q);       qb0.u4[1] = *(const uint4*)(q + 16);
        qb1.u4[0] = *(const uint4*)(q + 32);  qb1.u4[1] = *(const uint4*)(q + 48);
    }

    v8f ot[4];
#pragma unroll
    for (int t = 0; t < 4; ++t) ot[t] = (v8f){0, 0, 0, 0, 0, 0, 0, 0};
    float mrow = -1e30f, lrow = 0.f;

    // prologue: stage chunk 0 into buffer 0
    async_ld_b128(&smK[0][krow * H_D + kseg], Kp + (size_t)krow * H_D + kseg);
    async_ld_b128(&smV[0][vrow * 32 + vseg], Vp + (size_t)vrow * PIX + vseg);

    for (int it = 0; it < PIX / 32; ++it) {
        const int cur = it & 1;
        if (it + 1 < PIX / 32) {
            const int jb = (it + 1) * 32;
            async_ld_b128(&smK[1 - cur][krow * H_D + kseg],
                          Kp + (size_t)(jb + krow) * H_D + kseg);
            async_ld_b128(&smV[1 - cur][vrow * 32 + vseg],
                          Vp + (size_t)vrow * PIX + jb + vseg);
            wait_async_le2();        // previous stage (2 ops) landed, in order
        } else {
            wait_async_le0();
        }
        __syncthreads();             // stage `it` visible to every wave

        const _Float16* kt = &smK[cur][0];
        const _Float16* vt = &smV[cur][0];

        // ---- S^T (32 j-rows x 16 i-cols), K tiles as A-fragments from LDS --
        v8f s0 = {0, 0, 0, 0, 0, 0, 0, 0};
        v8f s1 = {0, 0, 0, 0, 0, 0, 0, 0};
        {
            const _Float16* k0 = kt + (size_t)n * H_D;          // tile0 row n
            const _Float16* k1 = kt + (size_t)(16 + n) * H_D;   // tile1 row n
            F16x16 a;
            a.u4[0] = *(const uint4*)(k0 + 8 * h);       a.u4[1] = *(const uint4*)(k0 + 16 + 8 * h);
            s0 = wmma_f16(a.v, qb0.v, s0);
            a.u4[0] = *(const uint4*)(k0 + 32 + 8 * h);  a.u4[1] = *(const uint4*)(k0 + 48 + 8 * h);
            s0 = wmma_f16(a.v, qb1.v, s0);
            a.u4[0] = *(const uint4*)(k1 + 8 * h);       a.u4[1] = *(const uint4*)(k1 + 16 + 8 * h);
            s1 = wmma_f16(a.v, qb0.v, s1);
            a.u4[0] = *(const uint4*)(k1 + 32 + 8 * h);  a.u4[1] = *(const uint4*)(k1 + 48 + 8 * h);
            s1 = wmma_f16(a.v, qb1.v, s1);
        }
        // ---- online softmax (per query i = n, replicated across h halves) --
        float mx = s0[0];
#pragma unroll
        for (int r = 0; r < 8; ++r) { mx = fmaxf(mx, s0[r]); mx = fmaxf(mx, s1[r]); }
        mx = fmaxf(mx, __shfl_xor(mx, 16, 32));
        const float mnew = fmaxf(mrow, mx);
        const float corr = __expf(mrow - mnew);
#pragma unroll
        for (int t = 0; t < 4; ++t)
#pragma unroll
            for (int r = 0; r < 8; ++r) ot[t][r] *= corr;
        float p0[8], p1[8], ls = 0.f;
#pragma unroll
        for (int r = 0; r < 8; ++r) {
            p0[r] = __expf(s0[r] - mnew);
            p1[r] = __expf(s1[r] - mnew);
            ls += p0[r] + p1[r];
        }
        lrow = lrow * corr + ls + __shfl_xor(ls, 16, 32);
        mrow = mnew;
        // ---- build P^T B-fragment via half-wave exchange -------------------
        F16x16 pb;
#pragma unroll
        for (int e = 0; e < 8; ++e) {
            float x0s = __shfl_xor(p0[e], 16, 32);
            float x1s = __shfl_xor(p1[e], 16, 32);
            pb.v[e]     = (_Float16)((h == 0) ? p0[e] : x1s);
            pb.v[e + 8] = (_Float16)((h == 0) ? x0s   : p1[e]);
        }
        // ---- O^T += V^T(d-block, 32 j) * P^T, V^T fragments from LDS -------
#pragma unroll
        for (int dblk = 0; dblk < 4; ++dblk) {
            const _Float16* vb = vt + (size_t)(dblk * 16 + n) * 32;
            F16x16 a;
            a.u4[0] = *(const uint4*)(vb + 16 * h);
            a.u4[1] = *(const uint4*)(vb + 16 * h + 8);
            ot[dblk] = wmma_f16(a.v, pb.v, ot[dblk]);
        }
        __syncthreads();             // everyone done reading buffer `cur`
    }

    // ---- normalize and store O[i, c=hh*64+d] as f16 (b,P,256) --------------
    const float inv = 1.0f / lrow;
    _Float16* outp = Of + ((size_t)b * PIX + i0 + n) * D_C + hh * H_D;
#pragma unroll
    for (int dblk = 0; dblk < 4; ++dblk) {
        F16x8 pk;
#pragma unroll
        for (int r = 0; r < 8; ++r) pk.h[r] = (_Float16)(ot[dblk][r] * inv);
        *(uint4*)(outp + dblk * 16 + 8 * h) = pk.u;   // rows d = dblk*16+8h+r
    }
}

// ---------------------------------------------------------------------------
extern "C" void kernel_launch(void* const* d_in, const int* in_sizes, int n_in,
                              void* d_out, int out_size, void* d_ws, size_t ws_size,
                              hipStream_t stream) {
    (void)in_sizes; (void)n_in; (void)out_size; (void)ws_size;
    const float* img  = (const float*)d_in[0];
    const float* gde  = (const float*)d_in[1];
    const float* ln1g = (const float*)d_in[2];
    const float* ln1b = (const float*)d_in[3];
    const float* ln2g = (const float*)d_in[4];
    const float* ln2b = (const float*)d_in[5];
    const float* qw1  = (const float*)d_in[6];
    const float* qb1  = (const float*)d_in[7];
    const float* qw2  = (const float*)d_in[8];
    const float* qb2  = (const float*)d_in[9];
    const float* kw1  = (const float*)d_in[10];
    const float* kb1  = (const float*)d_in[11];
    const float* kw2  = (const float*)d_in[12];
    const float* kb2  = (const float*)d_in[13];
    const float* vw1  = (const float*)d_in[14];
    const float* vb1  = (const float*)d_in[15];
    const float* vw2  = (const float*)d_in[16];
    const float* vb2  = (const float*)d_in[17];
    const float* ow   = (const float*)d_in[18];
    const float* ob   = (const float*)d_in[19];
    float* out = (float*)d_out;

    char* ws = (char*)d_ws;
    _Float16* XN = (_Float16*)(ws + (0ull  << 20));   // 4 MB  (B*P,256) f16
    _Float16* GN = (_Float16*)(ws + (4ull  << 20));   // 4 MB
    _Float16* WQ = (_Float16*)(ws + (8ull  << 20));   // 128 KB each
    _Float16* WK = WQ + 65536;
    _Float16* WV = WK + 65536;
    _Float16* WO = WV + 65536;
    float*    Y1 = (float*)(ws + (9ull  << 20));      // 8 MB  (B,256,P) f32
    _Float16* QF = (_Float16*)(ws + (17ull << 20));   // 4 MB  (BH,P,64)
    _Float16* KF = (_Float16*)(ws + (21ull << 20));   // 4 MB
    _Float16* VT = (_Float16*)(ws + (25ull << 20));   // 4 MB  (BH,64,P)
    _Float16* OF = (_Float16*)(ws + (29ull << 20));   // 4 MB  (B,P,256)

    const dim3 blk(256);
    // 1) weight conversion
    cvt_f16_kernel<<<256, blk, 0, stream>>>(qw1, WQ, 65536);
    cvt_f16_kernel<<<256, blk, 0, stream>>>(kw1, WK, 65536);
    cvt_f16_kernel<<<256, blk, 0, stream>>>(vw1, WV, 65536);
    cvt_f16_kernel<<<256, blk, 0, stream>>>(ow,  WO, 65536);
    // 2) layernorms
    lnorm_kernel<<<1024, blk, 0, stream>>>(img, ln1g, ln1b, XN);
    lnorm_kernel<<<1024, blk, 0, stream>>>(gde, ln2g, ln2b, GN);
    // 3/4) q,k,v: 1x1 GEMM then depthwise 3x3 (+ scale=hd^-0.5 folded into q)
    const dim3 ggrid(512, 2);
    gemm1x1_kernel<<<ggrid, blk, 0, stream>>>(XN, WQ, qb1, nullptr, Y1);
    dwconv_kernel<<<8192, blk, 0, stream>>>(Y1, qw2, qb2, 0.125f, 0, QF);
    gemm1x1_kernel<<<ggrid, blk, 0, stream>>>(GN, WK, kb1, nullptr, Y1);
    dwconv_kernel<<<8192, blk, 0, stream>>>(Y1, kw2, kb2, 1.0f, 0, KF);
    gemm1x1_kernel<<<ggrid, blk, 0, stream>>>(GN, WV, vb1, nullptr, Y1);
    dwconv_kernel<<<8192, blk, 0, stream>>>(Y1, vw2, vb2, 1.0f, 1, VT);
    // 5) attention (8 waves/block share one head's K/V via async LDS staging)
    attn_kernel<<<256, blk, 0, stream>>>(QF, KF, VT, OF);
    // 6) output projection + bias + residual
    gemm1x1_kernel<<<ggrid, blk, 0, stream>>>(OF, WO, ob, img, out);
}
